// VLSTMCell_11836929867860
// MI455X (gfx1250) — matrix operand order repaired and checked by reading, not compiled, not run on hardware
//
#include <hip/hip_runtime.h>
#include <hip/hip_bf16.h>

// ---------------------------------------------------------------------------
// VLSTM scan for MI455X (gfx1250), wave32 + WMMA bf16.
//
// Shapes: B=256, T=512, D=256, U=256, 4U=1024, UO=1.
// Decomposition: 16 workgroups, each owns 16 batch rows for all 512 steps.
// Per step, per WG: pre-act = [x_t | V] (16x512, bf16) @ Wstk (512x1024, bf16)
// via v_wmma_f32_16x16x32_bf16, f32 accumulate, + bias; then gates / scan
// update entirely in LDS + registers.
// ---------------------------------------------------------------------------

typedef __attribute__((ext_vector_type(16))) __bf16 v16bf;
typedef __attribute__((ext_vector_type(8)))  float  v8f;

#define KDIM   512   // D + U
#define NDIM   1024  // 4U
#define NTILES 64    // NDIM / 16
#define KTILES 16    // KDIM / 32
#define TSTEPS 512
#define UCOLS  256

__device__ __forceinline__ unsigned short f32_to_bf16_rne(float f) {
    unsigned u = __float_as_uint(f);
    u += 0x7FFFu + ((u >> 16) & 1u);
    return (unsigned short)(u >> 16);
}

// ---------------------------------------------------------------------------
// Prep: stack [kernel ; recurrent_kernel] (512 x 1024 f32) into bf16 B-matrix
// fragments laid out for B (32x16 tile) in contiguous k order, same as the A loads:
//   lanes 0-15: K = kt*32 + 0..15   (N = nt*16 + lane)
//   lanes 16-31: K = kt*32 + 16..31 (N = nt*16 + lane-16)
// Storage: wf[((nt*16 + kt)*32 + lane)*16 + j]  -> one v16bf load per lane.
// ---------------------------------------------------------------------------
__global__ void prep_w_bf16(const float* __restrict__ kern,   // [256][1024]
                            const float* __restrict__ rkern,  // [256][1024]
                            unsigned short* __restrict__ wf)  // [64][16][32][16]
{
    int idx = blockIdx.x * blockDim.x + threadIdx.x;
    if (idx >= NTILES * KTILES * 32 * 16) return;
    int j    =  idx        & 15;
    int lane = (idx >> 4)  & 31;
    int kt   = (idx >> 9)  & 15;
    int nt   =  idx >> 13;               // 0..63
    int k = kt * 32 + ((lane & 16) ? 16 : 0) + j;
    int n = nt * 16 + (lane & 15);
    float s = (k < 256) ? kern[k * NDIM + n] : rkern[(k - 256) * NDIM + n];
    wf[idx] = f32_to_bf16_rne(s);
}

// ---------------------------------------------------------------------------
// Fused scan kernel. grid = 16 WGs x 256 threads (8 waves).
// ---------------------------------------------------------------------------
__global__ __launch_bounds__(256, 1)
void vlstm_scan(const float* __restrict__ x,       // [256][512][256]
                const unsigned short* __restrict__ wf,  // bf16 fragments
                const float* __restrict__ bias,    // [1024]
                const float* __restrict__ dense,   // [256]
                float* __restrict__ out)           // [256][512]
{
    __shared__ unsigned short Arow[16][KDIM];   // bf16 [x_t | V], 16 KB
    __shared__ float          Gk[16][NDIM];     // f32 pre-activations, 64 KB
    __shared__ float          red[16][16];
    __shared__ float          h_sh[16];

    const int tid  = threadIdx.x;
    const int lane = tid & 31;
    const int wave = tid >> 5;             // 0..7 -> ntiles wave*8 .. +7
    const int r    = tid >> 4;             // 0..15: row for elementwise phase
    const int jb   = (tid & 15) << 4;      // 16-column slice base (0..240)
    const int rowbase = blockIdx.x * 16;   // global batch rows owned by WG

    // A-fragment per-lane decode (16-bit A 16x32 layout):
    //   lane<16: M=lane, K in kb+0..15 with kb=0
    //   lane>=16: M=lane-16, kb=16
    const int am = lane & 15;
    const int ak = (lane & 16) ? 16 : 0;
    // C/D (f32 16x16) layout: VGPR v -> M = v + (lane>=16 ? 8 : 0), N = lane&15
    const int gm = (lane & 16) ? 8 : 0;
    const int gn = lane & 15;

    float V[16];        // state slice, registers, persists across t
    float dk[16];       // dense_kernel slice
    #pragma unroll
    for (int q = 0; q < 16; ++q) { V[q] = 0.0f; dk[q] = dense[jb + q]; }
    #pragma unroll
    for (int q = 0; q < 16; ++q) Arow[r][UCOLS + jb + q] = 0;  // bf16(0) = V0

    for (int t = 0; t < TSTEPS; ++t) {
        // -------- phase 0: stage x_t (f32 -> bf16) into A row --------------
        {
            const float* xp = x + ((size_t)(rowbase + r) * TSTEPS + t) * 256 + jb;
            #pragma unroll
            for (int q = 0; q < 16; q += 4) {
                float4 v4 = *(const float4*)(xp + q);
                Arow[r][jb + q + 0] = f32_to_bf16_rne(v4.x);
                Arow[r][jb + q + 1] = f32_to_bf16_rne(v4.y);
                Arow[r][jb + q + 2] = f32_to_bf16_rne(v4.z);
                Arow[r][jb + q + 3] = f32_to_bf16_rne(v4.w);
            }
        }
        __syncthreads();  // A row complete (x part + V part from prev step)

        // -------- phase 1: [16,512] x [512,1024] bf16 WMMA, f32 accum ------
        v8f acc[8] = {};
        for (int kt = 0; kt < KTILES; ++kt) {
            union { v16bf v; float4 f4[2]; } af;
            const unsigned short* ap = &Arow[am][kt * 32 + ak];
            af.f4[0] = *(const float4*)(ap);        // K = kt*32 + ak + 0..7
            af.f4[1] = *(const float4*)(ap + 8);    // K = kt*32 + ak + 8..15
            #pragma unroll
            for (int i = 0; i < 8; ++i) {
                int nt = wave * 8 + i;
                v16bf bm = ((const v16bf*)wf)[(nt * KTILES + kt) * 32 + lane];
                acc[i] = __builtin_amdgcn_wmma_f32_16x16x32_bf16(
                    false, af.v, false, bm, (short)0, acc[i], false, false);
            }
        }
        // epilogue: + bias, scatter accumulators to LDS in C layout
        #pragma unroll
        for (int i = 0; i < 8; ++i) {
            int n = (wave * 8 + i) * 16 + gn;
            float bv = bias[n];
            #pragma unroll
            for (int v = 0; v < 8; ++v)
                Gk[gm + v][n] = acc[i][v] + bv;
        }
        __syncthreads();  // Gk complete

        // -------- phase 2: gates + per-row reduction -----------------------
        float Uc[16];
        float psum = 0.0f;
        #pragma unroll
        for (int q = 0; q < 16; ++q) {
            int j = jb + q;
            float gi = Gk[r][j];
            float gf = Gk[r][256 + j];
            float gc = Gk[r][512 + j];
            float go = Gk[r][768 + j];
            float ig = fminf(fmaxf(0.2f * gi + 0.5f, 0.0f), 1.0f);
            float fg = fminf(fmaxf(0.2f * gf + 0.5f, 0.0f), 1.0f);
            float cg = tanhf(gc);
            float og = fminf(fmaxf(0.2f * go + 0.5f, 0.0f), 1.0f);
            float vo = V[q];
            Uc[q] = fg * vo + ig * cg;
            psum += og * tanhf(vo) * dk[q];
        }
        red[r][tid & 15] = psum;
        __syncthreads();  // partials ready

        if (tid < 16) {
            float s = 0.0f;
            #pragma unroll
            for (int q = 0; q < 16; ++q) s += red[tid][q];
            float h = 1.0f / (1.0f + __expf(-s));
            h_sh[tid] = h;
            out[(size_t)(rowbase + tid) * TSTEPS + t] = h;
        }
        __syncthreads();  // h ready

        // -------- phase 3: state update (registers + bf16 copy to A) -------
        float h = h_sh[r];
        #pragma unroll
        for (int q = 0; q < 16; ++q) {
            float vn = h * Uc[q] + (1.0f - h) * V[q];
            V[q] = vn;
            Arow[r][UCOLS + jb + q] = f32_to_bf16_rne(vn);
        }
        // next iteration's barrier (after phase 0) covers these writes
    }
}

extern "C" void kernel_launch(void* const* d_in, const int* in_sizes, int n_in,
                              void* d_out, int out_size, void* d_ws, size_t ws_size,
                              hipStream_t stream) {
    const float* inputs = (const float*)d_in[0];   // [256,512,256]
    const float* kern   = (const float*)d_in[1];   // [256,1024]
    const float* rkern  = (const float*)d_in[2];   // [256,1024]
    const float* dense  = (const float*)d_in[3];   // [256,1]
    const float* bias   = (const float*)d_in[4];   // [1024]

    unsigned short* wf = (unsigned short*)d_ws;    // 512*1024*2 = 1 MB scratch

    const int welems = NTILES * KTILES * 32 * 16;  // 524288
    prep_w_bf16<<<(welems + 255) / 256, 256, 0, stream>>>(kern, rkern, wf);
    vlstm_scan<<<16, 256, 0, stream>>>(inputs, wf, bias, dense, (float*)d_out);
}
